// PRXBlock_1709396984589
// MI455X (gfx1250) — matrix operand order repaired and checked by reading, not compiled, not run on hardware
//
#include <hip/hip_runtime.h>

typedef __attribute__((ext_vector_type(16))) _Float16 v16h;
typedef __attribute__((ext_vector_type(8)))  float    v8f;
typedef unsigned int uint;

// ---------------------------------------------------------------------------
// Problem constants (B=1 fixed by the reference setup)
// ---------------------------------------------------------------------------
#define LI   2048
#define LT   256
#define LKV  2304          // LT + LI
#define DMODEL 1536
#define DFF  6144
#define NH   24
#define DH   64

// ---------------------------------------------------------------------------
// WMMA fragment gathers (layouts per CDNA5 ISA 7.12.2, f16 16x16x32)
// ---------------------------------------------------------------------------
// A (16x32, row-major source with ldHalf halves per row):
//   lane l: m = mBase + (l&15); kb = 8*(l>>4)
//   half t<8 : k = kb + t ; half t>=8 : k = kb + 16 + (t-8)
__device__ inline v16h frag_a_rowmajor(const _Float16* base, int ldHalf,
                                       int lane, int mBase, int kBase) {
  const uint* bd = (const uint*)base;
  int ldd = ldHalf >> 1;
  int m  = mBase + (lane & 15);
  int kb = kBase + ((lane >> 4) << 3);
  union { v16h h; uint u[8]; } f;
  int r0 = m * ldd + (kb >> 1);
  #pragma unroll
  for (int i = 0; i < 4; i++) f.u[i] = bd[r0 + i];
  int r1 = m * ldd + ((kb + 16) >> 1);
  #pragma unroll
  for (int i = 0; i < 4; i++) f.u[4 + i] = bd[r1 + i];
  return f.h;
}

// B (32x16) from a K-major tile (tile[k][n], ldHalf halves per k-row):
//   lane l: n = nBase + (l&15); kb = kBase + 8*(l>>4)
//   half t<8 : k = kb + t ; half t>=8 : k = kb + 16 + (t-8)
__device__ inline v16h frag_b_kmajor(const _Float16* base, int ldHalf,
                                     int lane, int kBase, int nBase) {
  union { v16h h; _Float16 e[16]; } f;
  int n  = nBase + (lane & 15);
  int kb = kBase + ((lane >> 4) << 3);
  #pragma unroll
  for (int i = 0; i < 8; i++) {
    f.e[i]     = base[(kb + i) * ldHalf + n];
    f.e[8 + i] = base[(kb + 16 + i) * ldHalf + n];
  }
  return f.h;
}

__device__ inline float gelu_tanh(float x) {
  float x3 = x * x * x;
  float t  = tanhf(0.7978845608028654f * (x + 0.044715f * x3));
  return 0.5f * x * (1.0f + t);
}

// ---------------------------------------------------------------------------
// Small kernels
// ---------------------------------------------------------------------------

// mod = silu(vec) @ W_mod^T + b_mod      ([6*D] outputs, K = D)
__global__ __launch_bounds__(256)
void mod_kernel(const float* __restrict__ vec, const float* __restrict__ Wm,
                const float* __restrict__ bm, float* __restrict__ mod) {
  __shared__ float sv[DMODEL];
  for (int i = threadIdx.x; i < DMODEL; i += 256) {
    float v = vec[i];
    sv[i] = v / (1.0f + __expf(-v));
  }
  __syncthreads();
  int o = blockIdx.x * 256 + threadIdx.x;           // < 6*DMODEL
  const float* w = Wm + (size_t)o * DMODEL;
  float acc = bm[o];
  for (int j = 0; j < DMODEL; j++) acc += sv[j] * w[j];
  mod[o] = acc;
}

__global__ __launch_bounds__(256)
void bias_kernel(const unsigned char* __restrict__ mask, float* __restrict__ bias) {
  int i = blockIdx.x * 256 + threadIdx.x;
  if (i < LKV) bias[i] = (i < LT) ? (mask[i] ? 0.0f : -1.0e30f) : 0.0f;
}

__global__ __launch_bounds__(256)
void cvt_f32_f16(const float* __restrict__ in, _Float16* __restrict__ out, int n) {
  int i = blockIdx.x * 256 + threadIdx.x;
  if (i < n) out[i] = (_Float16)in[i];
}

// out_h[row,:] = f16( (1+mod[scaleOff+c]) * LN(x[row,:]) + mod[shiftOff+c] )
__global__ __launch_bounds__(256)
void ln_mod_kernel(const float* __restrict__ X, const float* __restrict__ mod,
                   int shiftOff, int scaleOff, _Float16* __restrict__ outH) {
  int row = blockIdx.x;
  int tid = threadIdx.x, lane = tid & 31, wave = tid >> 5;
  const float* x = X + (size_t)row * DMODEL;
  __shared__ float red[8];
  float lv[6];
  float s = 0.0f;
  #pragma unroll
  for (int i = 0; i < 6; i++) { lv[i] = x[tid + i * 256]; s += lv[i]; }
  #pragma unroll
  for (int off = 16; off; off >>= 1) s += __shfl_xor(s, off, 32);
  if (lane == 0) red[wave] = s;
  __syncthreads();
  float mean = 0.0f;
  #pragma unroll
  for (int i = 0; i < 8; i++) mean += red[i];
  mean *= (1.0f / (float)DMODEL);
  __syncthreads();
  float vs = 0.0f;
  #pragma unroll
  for (int i = 0; i < 6; i++) { float d = lv[i] - mean; vs += d * d; }
  #pragma unroll
  for (int off = 16; off; off >>= 1) vs += __shfl_xor(vs, off, 32);
  if (lane == 0) red[wave] = vs;
  __syncthreads();
  float var = 0.0f;
  #pragma unroll
  for (int i = 0; i < 8; i++) var += red[i];
  var *= (1.0f / (float)DMODEL);
  float rstd = rsqrtf(var + 1e-6f);
  #pragma unroll
  for (int i = 0; i < 6; i++) {
    int c = tid + i * 256;
    float v = (lv[i] - mean) * rstd;
    outH[(size_t)row * DMODEL + c] =
        (_Float16)((1.0f + mod[scaleOff + c]) * v + mod[shiftOff + c]);
  }
}

// Per-(l,h): rmsnorm q/k over DH, scale, rope; pack f16 q/k/v in [H][L][DH]
__global__ __launch_bounds__(64)
void qkprep_img(const float* __restrict__ qkv, const float* __restrict__ pe,
                const float* __restrict__ q_scale, const float* __restrict__ k_scale,
                _Float16* __restrict__ qh, _Float16* __restrict__ kh,
                _Float16* __restrict__ vh) {
  int l = blockIdx.x, h = blockIdx.y, d = threadIdx.x;
  const float* base = qkv + (size_t)l * (3 * DMODEL) + h * DH;
  float q = base[d], k = base[DMODEL + d], v = base[2 * DMODEL + d];
  __shared__ float rq[64], rk[64], nq[64], nk[64];
  rq[d] = q * q; rk[d] = k * k;
  __syncthreads();
  for (int off = 32; off; off >>= 1) {
    if (d < off) { rq[d] += rq[d + off]; rk[d] += rk[d + off]; }
    __syncthreads();
  }
  float rrq = rsqrtf(rq[0] * (1.0f / DH) + 1e-6f);
  float rrk = rsqrtf(rk[0] * (1.0f / DH) + 1e-6f);
  nq[d] = q * rrq * q_scale[d];
  nk[d] = k * rrk * k_scale[d];
  __syncthreads();
  int d2 = d >> 1, j = d & 1;
  const float* p = pe + ((((size_t)l * (DH / 2) + d2) * 2 + j) * 2);
  float qr = p[0] * nq[2 * d2] + p[1] * nq[2 * d2 + 1];
  float kr = p[0] * nk[2 * d2] + p[1] * nk[2 * d2 + 1];
  qh[((size_t)h * LI + l) * DH + d]        = (_Float16)qr;
  kh[((size_t)h * LKV + LT + l) * DH + d]  = (_Float16)kr;
  vh[((size_t)h * LKV + LT + l) * DH + d]  = (_Float16)v;
}

__global__ __launch_bounds__(64)
void qkprep_txt(const float* __restrict__ tkv, const float* __restrict__ tks,
                _Float16* __restrict__ kh, _Float16* __restrict__ vh) {
  int l = blockIdx.x, h = blockIdx.y, d = threadIdx.x;
  const float* base = tkv + (size_t)l * (2 * DMODEL) + h * DH;
  float k = base[d], v = base[DMODEL + d];
  __shared__ float rk[64];
  rk[d] = k * k;
  __syncthreads();
  for (int off = 32; off; off >>= 1) {
    if (d < off) rk[d] += rk[d + off];
    __syncthreads();
  }
  float rrk = rsqrtf(rk[0] * (1.0f / DH) + 1e-6f);
  kh[((size_t)h * LKV + l) * DH + d] = (_Float16)(k * rrk * tks[d]);
  vh[((size_t)h * LKV + l) * DH + d] = (_Float16)v;
}

// ---------------------------------------------------------------------------
// Generic f16 WMMA GEMM:  C[M,N] = Xh[M,K] @ W[N,K]^T  (W is f32, row-major)
// Block tile 128x128, K-step 32, 8 waves (4M x 2N), wave tile 32x64.
// 2-stage pipeline: tile t+1 is prefetched into VGPRs (global loads in
// flight) while tile t is computed from LDS; regs -> other LDS buffer after
// the barrier. Global latency overlaps the WMMA/DS compute phase.
// Epilogue MODE: 0 store f32 | 1 store f16(gelu) | 2 store f16(v*aux)
//               3 outF = res + gateVec[n]*v | 4 outF += gateVec[n]*v
// ---------------------------------------------------------------------------
#define LDA 34   // halves per As row (17 dwords, coprime bank stride)
#define LDB 130  // halves per Bs row (65 dwords)

template <int MODE>
__global__ __launch_bounds__(256)
void gemm_f16(const _Float16* __restrict__ X, const float* __restrict__ W,
              int M, int N, int K,
              float* __restrict__ outF, _Float16* __restrict__ outH,
              const _Float16* __restrict__ auxH, const float* __restrict__ resF,
              const float* __restrict__ gateVec) {
  __shared__ _Float16 As[2][128 * LDA];
  __shared__ _Float16 Bs[2][32 * LDB];

  int tid  = threadIdx.x;
  int lane = tid & 31, wave = tid >> 5;
  int wm = wave >> 1, wn = wave & 1;
  int nb = blockIdx.x * 128, mb = blockIdx.y * 128;

  v8f zero8 = {0.f, 0.f, 0.f, 0.f, 0.f, 0.f, 0.f, 0.f};
  v8f acc[2][4];
  #pragma unroll
  for (int i = 0; i < 2; i++)
    #pragma unroll
    for (int j = 0; j < 4; j++) acc[i][j] = zero8;

  int r   = tid >> 1;       // 0..127
  int seg = tid & 1;        // 0..1
  uint  aReg[8];
  float bReg[16];

  const uint*  Xrow = (const uint*)(X + (size_t)(mb + r) * K + seg * 16);
  const float* Wrow = W + (size_t)(nb + r) * K + seg * 16;

  auto stage_regs = [&](int k0) {
    const uint* Xu = Xrow + (k0 >> 1);
    #pragma unroll
    for (int i = 0; i < 8; i++) aReg[i] = Xu[i];
    const float* Wp = Wrow + k0;
    __builtin_prefetch((const void*)(Wp + 32), 0, 0);
    #pragma unroll
    for (int i = 0; i < 16; i++) bReg[i] = Wp[i];
  };
  auto write_lds = [&](int buf) {
    uint* Ad = (uint*)As[buf];
    int b = r * (LDA / 2) + seg * 8;
    #pragma unroll
    for (int i = 0; i < 8; i++) Ad[b + i] = aReg[i];
    #pragma unroll
    for (int i = 0; i < 16; i++)
      Bs[buf][(seg * 16 + i) * LDB + r] = (_Float16)bReg[i];
  };

  // prologue: tile 0 staged to buffer 0
  stage_regs(0);
  write_lds(0);
  __syncthreads();

  int T = K >> 5;
  for (int t = 0; t < T; t++) {
    int cur = t & 1;
    if (t + 1 < T) stage_regs((t + 1) << 5);   // global loads in flight

    v16h a0 = frag_a_rowmajor(As[cur], LDA, lane, wm * 32, 0);
    v16h a1 = frag_a_rowmajor(As[cur], LDA, lane, wm * 32 + 16, 0);
    #pragma unroll
    for (int j = 0; j < 4; j++) {
      v16h b = frag_b_kmajor(Bs[cur], LDB, lane, 0, wn * 64 + j * 16);
      acc[0][j] = __builtin_amdgcn_wmma_f32_16x16x32_f16(
          false, a0, false, b, (short)0, acc[0][j], false, false);
      acc[1][j] = __builtin_amdgcn_wmma_f32_16x16x32_f16(
          false, a1, false, b, (short)0, acc[1][j], false, false);
    }

    if (t + 1 < T) {
      __syncthreads();            // everyone done reading As/Bs[cur]
      write_lds(cur ^ 1);
      __syncthreads();            // next tile visible to all waves
    }
  }

  // --- epilogue (C layout: lane n = l&15, vgpr r -> m = r + 8*(l>>4)) ---
  int nloc = lane & 15;
  int moff = (lane >> 4) << 3;
  #pragma unroll
  for (int i = 0; i < 2; i++) {
    #pragma unroll
    for (int j = 0; j < 4; j++) {
      int mRow = mb + wm * 32 + i * 16 + moff;
      int nCol = nb + wn * 64 + j * 16 + nloc;
      #pragma unroll
      for (int rr = 0; rr < 8; rr++) {
        size_t idx = (size_t)(mRow + rr) * N + nCol;
        float v = acc[i][j][rr];
        if (MODE == 0) outF[idx] = v;
        else if (MODE == 1) outH[idx] = (_Float16)gelu_tanh(v);
        else if (MODE == 2) outH[idx] = (_Float16)(v * (float)auxH[idx]);
        else if (MODE == 3) outF[idx] = resF[idx] + gateVec[nCol] * v;
        else                outF[idx] = outF[idx] + gateVec[nCol] * v;
      }
    }
  }
}

// ---------------------------------------------------------------------------
// Flash attention: block = 4 waves, one head, 64 q-rows (16 per wave).
// K-loop in steps of 64 keys; S = Q K^T and O += P V both via WMMA f16.
// ---------------------------------------------------------------------------
#define LDK 66  // halves per row in Ks/Vs/Ps tiles (33 dwords)

__global__ __launch_bounds__(128)
void attn_kernel(const _Float16* __restrict__ qh, const _Float16* __restrict__ kh,
                 const _Float16* __restrict__ vh, const float* __restrict__ bias,
                 _Float16* __restrict__ attnh) {
  int h = blockIdx.y;
  int qbase = blockIdx.x * 64;
  int tid = threadIdx.x, wave = tid >> 5, lane = tid & 31;

  __shared__ _Float16 Ks[DH * LDK];        // [dh][key]  (transposed)
  __shared__ _Float16 Vs[64 * LDK];        // [key][dh]
  __shared__ _Float16 Ps[4][16 * LDK];     // per-wave P scratch
  __shared__ float    sbias[LKV];

  for (int i = tid; i < LKV; i += 128) sbias[i] = bias[i];

  // Q fragments (kept in registers for the whole K loop)
  const _Float16* qrow = qh + ((size_t)h * LI + qbase + wave * 16) * DH;
  v16h qa0 = frag_a_rowmajor(qrow, DH, lane, 0, 0);
  v16h qa1 = frag_a_rowmajor(qrow, DH, lane, 0, 32);

  v8f zero8 = {0.f, 0.f, 0.f, 0.f, 0.f, 0.f, 0.f, 0.f};
  v8f oAcc[4];
  #pragma unroll
  for (int nt = 0; nt < 4; nt++) oAcc[nt] = zero8;
  float rM[8], rL[8];
  #pragma unroll
  for (int rr = 0; rr < 8; rr++) { rM[rr] = -1.0e30f; rL[rr] = 0.0f; }

  const float scale = 0.125f;  // DH^-0.5
  int nloc = lane & 15;
  int moff = (lane >> 4) << 3;

  __syncthreads();

  for (int kb0 = 0; kb0 < LKV; kb0 += 64) {
    // cooperative stage of K (transposed) and V (natural)
    {
      int key = tid >> 1, seg = tid & 1;
      const _Float16* kp = kh + ((size_t)h * LKV + kb0 + key) * DH + seg * 32;
      #pragma unroll
      for (int d = 0; d < 32; d++) Ks[(seg * 32 + d) * LDK + key] = kp[d];
      const uint* vp = (const uint*)(vh + ((size_t)h * LKV + kb0 + key) * DH + seg * 32);
      uint* vd = (uint*)&Vs[key * LDK + seg * 32];
      #pragma unroll
      for (int i = 0; i < 16; i++) vd[i] = vp[i];
    }
    __syncthreads();

    // S = Q K^T (4 key-tiles x 2 K-steps)
    v8f s[4];
    #pragma unroll
    for (int nt = 0; nt < 4; nt++) s[nt] = zero8;
    #pragma unroll
    for (int nt = 0; nt < 4; nt++) {
      v16h b0 = frag_b_kmajor(Ks, LDK, lane, 0, nt * 16);
      s[nt] = __builtin_amdgcn_wmma_f32_16x16x32_f16(false, qa0, false, b0,
                                                     (short)0, s[nt], false, false);
      v16h b1 = frag_b_kmajor(Ks, LDK, lane, 32, nt * 16);
      s[nt] = __builtin_amdgcn_wmma_f32_16x16x32_f16(false, qa1, false, b1,
                                                     (short)0, s[nt], false, false);
    }

    // scale + bias
    float bv[4];
    #pragma unroll
    for (int nt = 0; nt < 4; nt++) bv[nt] = sbias[kb0 + nt * 16 + nloc];
    #pragma unroll
    for (int nt = 0; nt < 4; nt++)
      #pragma unroll
      for (int rr = 0; rr < 8; rr++) s[nt][rr] = s[nt][rr] * scale + bv[nt];

    // online softmax update
    float alpha[8];
    #pragma unroll
    for (int rr = 0; rr < 8; rr++) {
      float mx = s[0][rr];
      #pragma unroll
      for (int nt = 1; nt < 4; nt++) mx = fmaxf(mx, s[nt][rr]);
      #pragma unroll
      for (int off = 1; off < 16; off <<= 1) mx = fmaxf(mx, __shfl_xor(mx, off, 32));
      float nm = fmaxf(rM[rr], mx);
      alpha[rr] = __expf(rM[rr] - nm);
      rM[rr] = nm;
    }
    #pragma unroll
    for (int rr = 0; rr < 8; rr++) {
      float sum = 0.0f;
      #pragma unroll
      for (int nt = 0; nt < 4; nt++) {
        float p = __expf(s[nt][rr] - rM[rr]);
        s[nt][rr] = p;
        sum += p;
      }
      #pragma unroll
      for (int off = 1; off < 16; off <<= 1) sum += __shfl_xor(sum, off, 32);
      rL[rr] = rL[rr] * alpha[rr] + sum;
      oAcc[0][rr] *= alpha[rr]; oAcc[1][rr] *= alpha[rr];
      oAcc[2][rr] *= alpha[rr]; oAcc[3][rr] *= alpha[rr];
    }

    // P (C layout) -> LDS -> A layout
    #pragma unroll
    for (int nt = 0; nt < 4; nt++)
      #pragma unroll
      for (int rr = 0; rr < 8; rr++)
        Ps[wave][(moff + rr) * LDK + nt * 16 + nloc] = (_Float16)s[nt][rr];
    // same-wave DS ops are in-order: reads below observe the writes above

    #pragma unroll
    for (int ks = 0; ks < 2; ks++) {
      v16h pa = frag_a_rowmajor(Ps[wave], LDK, lane, 0, ks * 32);
      #pragma unroll
      for (int nt = 0; nt < 4; nt++) {
        v16h vb = frag_b_kmajor(Vs, LDK, lane, ks * 32, nt * 16);
        oAcc[nt] = __builtin_amdgcn_wmma_f32_16x16x32_f16(
            false, pa, false, vb, (short)0, oAcc[nt], false, false);
      }
    }
    __syncthreads();
  }

  // finalize: attn[l, h*64 + d] in f16
  #pragma unroll
  for (int nt = 0; nt < 4; nt++) {
    #pragma unroll
    for (int rr = 0; rr < 8; rr++) {
      int qr = qbase + wave * 16 + moff + rr;
      float inv = 1.0f / fmaxf(rL[rr], 1e-30f);
      attnh[(size_t)qr * DMODEL + h * DH + nt * 16 + nloc] =
          (_Float16)(oAcc[nt][rr] * inv);
    }
  }
}

// ---------------------------------------------------------------------------
// Host-side orchestration
// ---------------------------------------------------------------------------
extern "C" void kernel_launch(void* const* d_in, const int* in_sizes, int n_in,
                              void* d_out, int out_size, void* d_ws, size_t ws_size,
                              hipStream_t stream) {
  const float* img       = (const float*)d_in[0];
  const float* txt       = (const float*)d_in[1];
  const float* vec       = (const float*)d_in[2];
  const float* pe        = (const float*)d_in[3];
  const unsigned char* m = (const unsigned char*)d_in[4];
  const float* W_mod     = (const float*)d_in[5];
  const float* b_mod     = (const float*)d_in[6];
  const float* W_qkv     = (const float*)d_in[7];
  const float* q_scale   = (const float*)d_in[8];
  const float* k_scale   = (const float*)d_in[9];
  const float* tk_scale  = (const float*)d_in[10];
  const float* W_txt_kv  = (const float*)d_in[11];
  const float* W_attn    = (const float*)d_in[12];
  const float* W_gate    = (const float*)d_in[13];
  const float* W_up      = (const float*)d_in[14];
  const float* W_down    = (const float*)d_in[15];
  float* out = (float*)d_out;

  char* w = (char*)d_ws;
  auto carve = [&](size_t bytes) -> char* {
    char* p = w;
    w += (bytes + 255) & ~(size_t)255;
    return p;
  };
  float*    mod  = (float*)carve((size_t)6 * DMODEL * 4);
  float*    bias = (float*)carve((size_t)LKV * 4);
  _Float16* xh   = (_Float16*)carve((size_t)LI * DMODEL * 2);
  _Float16* txth = (_Float16*)carve((size_t)LT * DMODEL * 2);
  float*    qkv  = (float*)carve((size_t)LI * 3 * DMODEL * 4);
  float*    tkv  = (float*)carve((size_t)LT * 2 * DMODEL * 4);
  _Float16* qh   = (_Float16*)carve((size_t)NH * LI * DH * 2);
  _Float16* kh   = (_Float16*)carve((size_t)NH * LKV * DH * 2);
  _Float16* vh   = (_Float16*)carve((size_t)NH * LKV * DH * 2);
  _Float16* ah   = (_Float16*)carve((size_t)LI * DMODEL * 2);
  _Float16* yh   = (_Float16*)carve((size_t)LI * DMODEL * 2);
  _Float16* gh   = (_Float16*)carve((size_t)LI * DFF * 2);
  _Float16* hh   = (_Float16*)carve((size_t)LI * DFF * 2);

  // modulation + mask bias + txt f16
  mod_kernel<<<(6 * DMODEL) / 256, 256, 0, stream>>>(vec, W_mod, b_mod, mod);
  bias_kernel<<<(LKV + 255) / 256, 256, 0, stream>>>(m, bias);
  cvt_f32_f16<<<(LT * DMODEL) / 256, 256, 0, stream>>>(txt, txth, LT * DMODEL);

  // attention branch
  ln_mod_kernel<<<LI, 256, 0, stream>>>(img, mod, 0, DMODEL, xh);
  gemm_f16<0><<<dim3((3 * DMODEL) / 128, LI / 128), 256, 0, stream>>>(
      xh, W_qkv, LI, 3 * DMODEL, DMODEL, qkv, nullptr, nullptr, nullptr, nullptr);
  gemm_f16<0><<<dim3((2 * DMODEL) / 128, LT / 128), 256, 0, stream>>>(
      txth, W_txt_kv, LT, 2 * DMODEL, DMODEL, tkv, nullptr, nullptr, nullptr, nullptr);
  qkprep_img<<<dim3(LI, NH), 64, 0, stream>>>(qkv, pe, q_scale, k_scale, qh, kh, vh);
  qkprep_txt<<<dim3(LT, NH), 64, 0, stream>>>(tkv, tk_scale, kh, vh);
  attn_kernel<<<dim3(LI / 64, NH), 128, 0, stream>>>(qh, kh, vh, bias, ah);
  gemm_f16<3><<<dim3(DMODEL / 128, LI / 128), 256, 0, stream>>>(
      ah, W_attn, LI, DMODEL, DMODEL, out, nullptr, nullptr, img, mod + 2 * DMODEL);

  // MLP branch
  ln_mod_kernel<<<LI, 256, 0, stream>>>(out, mod, 3 * DMODEL, 4 * DMODEL, yh);
  gemm_f16<1><<<dim3(DFF / 128, LI / 128), 256, 0, stream>>>(
      yh, W_gate, LI, DFF, DMODEL, nullptr, gh, nullptr, nullptr, nullptr);
  gemm_f16<2><<<dim3(DFF / 128, LI / 128), 256, 0, stream>>>(
      yh, W_up, LI, DFF, DMODEL, nullptr, hh, gh, nullptr, nullptr);
  gemm_f16<4><<<dim3(DMODEL / 128, LI / 128), 256, 0, stream>>>(
      hh, W_down, LI, DMODEL, DFF, out, nullptr, nullptr, nullptr, mod + 5 * DMODEL);
}
